// CausalAttention_49709951484070
// MI455X (gfx1250) — matrix operand order repaired and checked by reading, not compiled, not run on hardware
//
#include <hip/hip_runtime.h>
#include <hip/hip_bf16.h>
#include <math.h>

// ---------------------------------------------------------------------------
// Causal attention, MI455X (gfx1250, wave32, WMMA 16x16x32 f16 -> f32 acc)
//   X:[4,2048,1024] f32, Wq/Wk/Wv:[1024,1024] f32 -> out:[4,2048,1024] f32
// 4-stage pipeline: QKV projection -> QK^T (scale+causal) -> softmax -> P*V
// Double-buffered LDS, 1 barrier per K-step, async global->LDS where the
// tile is a pure f16 copy (ASYNCcnt path), register-staged where conversion
// or transpose is needed.
// Workspace: Qh/Kh/Vh f16 (48MB) + S f32 (64MB) + P f16 (32MB) = 144MB
// ---------------------------------------------------------------------------

typedef __attribute__((ext_vector_type(16))) _Float16 v16h;
typedef __attribute__((ext_vector_type(8)))  _Float16 h8;
typedef __attribute__((ext_vector_type(4)))  _Float16 h4;
typedef __attribute__((ext_vector_type(8)))  float    v8f;
typedef __attribute__((ext_vector_type(4)))  int      i4;

union Frag16 { v16h v; h8 g[2]; };

#define SEQ    2048
#define DMODEL 1024
#define NBATCH 4
#define ROWS   8192            // NBATCH * SEQ (flattened batch for projections)

#define M_BLK  128             // block tile rows
#define N_BLK  64              // block tile cols
#define K_STEP 32              // WMMA K per iteration
#define PA     40              // LDS A-tile pitch in halfs (80B, 16B aligned)
#define PB     40              // LDS B-tile pitch in halfs

#define NEG_INF (-__builtin_inff())

// ---- CDNA5 async global->LDS (ASYNCcnt) with compile-safe fallback --------
#if defined(__has_builtin)
#if __has_builtin(__builtin_amdgcn_global_load_async_to_lds_b128)
#define HAVE_ASYNC_LDS 1
#endif
#endif
#ifndef HAVE_ASYNC_LDS
#define HAVE_ASYNC_LDS 0
#endif

__device__ __forceinline__ void wait_async0() {
#if defined(__has_builtin) && __has_builtin(__builtin_amdgcn_s_wait_asynccnt)
    __builtin_amdgcn_s_wait_asynccnt(0);
#else
    asm volatile("s_wait_asynccnt 0x0" ::: "memory");
#endif
}

#if HAVE_ASYNC_LDS
// Builtin signature (from toolchain diagnostic): param0 = AS(1) int4*,
// param1 = AS(3) int4*, then imm offset + imm cpol.
__device__ __forceinline__ void async_copy_b128(const _Float16* g, _Float16* l) {
    __builtin_amdgcn_global_load_async_to_lds_b128(
        (__attribute__((address_space(1))) i4*)(g),
        (__attribute__((address_space(3))) i4*)(l), 0, 0);
}
#endif

// D = A(16x32 f16) x B(32x16 f16) + C(f32)
__device__ __forceinline__ v8f wmma_f16(v16h a, v16h b, v8f c) {
    return __builtin_amdgcn_wmma_f32_16x16x32_f16(
        false, a, false, b, (short)0, c, false, false);
}

// A fragment: lane holds row M = lane&15; elements e<8 -> K=(lane>>4)*8+e,
// e>=8 -> K=16+(lane>>4)*8+(e-8). Two contiguous ds_load_b128.
__device__ __forceinline__ void load_a_frag(Frag16& f, const _Float16* rowBase,
                                            int lane) {
    const _Float16* p = rowBase + (lane >> 4) * 8;
    f.g[0] = *(const h8*)(p);
    f.g[1] = *(const h8*)(p + 16);
}

// B fragment from transposed LDS tile Bt[n][k]: lane holds column N=lane&15,
// e<8 -> K=(lane>>4)*8+e, e>=8 -> K=16+(lane>>4)*8+(e-8) (two ds_load_b128).
__device__ __forceinline__ void load_b_frag(Frag16& f, const _Float16* bt,
                                            int nRow, int lane) {
    const _Float16* p = bt + (size_t)nRow * PB + (lane >> 4) * 8;
    f.g[0] = *(const h8*)(p);
    f.g[1] = *(const h8*)(p + 16);
}

// one wave: 32x32 tile = 2 A-frags x 2 B-frags -> 4 WMMA
__device__ __forceinline__ void compute_tile(v8f acc[2][2], const _Float16* bufA,
                                             const _Float16* bufB, int wm, int wn,
                                             int lane) {
    const int rIT = lane & 15;
    Frag16 bf[2];
    load_b_frag(bf[0], bufB, wn + rIT,      lane);
    load_b_frag(bf[1], bufB, wn + 16 + rIT, lane);
#pragma unroll
    for (int mi = 0; mi < 2; ++mi) {
        Frag16 af;
        load_a_frag(af, bufA + (size_t)(wm + mi * 16 + rIT) * PA, lane);
        acc[mi][0] = wmma_f16(af.v, bf[0].v, acc[mi][0]);
        acc[mi][1] = wmma_f16(af.v, bf[1].v, acc[mi][1]);
    }
}

// ---------------------------------------------------------------------------
// Kernel 1: Qh/Kh/Vh = f16( X @ W{q,k,v} ).  grid = (64 Mtiles, 16 Ntiles, 3)
// Register-staged (needs f32->f16 conversion), double-buffered, pipelined.
// ---------------------------------------------------------------------------
__global__ __launch_bounds__(256)
void qkv_gemm(const float* __restrict__ X,
              const float* __restrict__ Wq, const float* __restrict__ Wk,
              const float* __restrict__ Wv,
              _Float16* __restrict__ Qh, _Float16* __restrict__ Kh,
              _Float16* __restrict__ Vh)
{
    __shared__ _Float16 lsA[2][M_BLK * PA];
    __shared__ _Float16 lsB[2][N_BLK * PB];

    const int tid  = threadIdx.x;
    const int lane = tid & 31;
    const int wave = tid >> 5;
    const int m0   = blockIdx.x * M_BLK;
    const int n0   = blockIdx.y * N_BLK;
    const float* W   = (blockIdx.z == 0) ? Wq : (blockIdx.z == 1) ? Wk : Wv;
    _Float16*    Out = (blockIdx.z == 0) ? Qh : (blockIdx.z == 1) ? Kh : Vh;
    const int wm = (wave >> 1) * 32;       // 4x2 wave grid, 32x32 per wave
    const int wn = (wave & 1) * 32;

    // per-thread staging geometry (constant): A: 4 float4 chunks, rows
    // tid>>3 + i*32, col (tid&7)*4; B: 2 float4 chunks, k rows tid>>4 + i*16,
    // n cols (tid&15)*4 (transposed into LDS).
    const int aRow = tid >> 3, aCol = (tid & 7) * 4;
    const int bRow = tid >> 4, bCol = (tid & 15) * 4;
    const float* aG = X + (size_t)(m0 + aRow) * DMODEL + aCol;
    const float* bG = W + (size_t)bRow * DMODEL + n0 + bCol;

    float4 xa[4], wb[2];
    auto stage_load = [&](int k0) {
#pragma unroll
        for (int i = 0; i < 4; ++i)
            xa[i] = *(const float4*)(aG + k0 + (size_t)i * 32 * DMODEL);
#pragma unroll
        for (int i = 0; i < 2; ++i)
            wb[i] = *(const float4*)(bG + (size_t)(k0 + i * 16) * DMODEL);
    };
    auto stage_store = [&](int buf) {
#pragma unroll
        for (int i = 0; i < 4; ++i) {
            h4 hx = { (_Float16)xa[i].x, (_Float16)xa[i].y,
                      (_Float16)xa[i].z, (_Float16)xa[i].w };
            *(h4*)(&lsA[buf][(aRow + i * 32) * PA + aCol]) = hx;
        }
#pragma unroll
        for (int i = 0; i < 2; ++i) {
            int kk = bRow + i * 16;
            lsB[buf][(bCol + 0) * PB + kk] = (_Float16)wb[i].x;
            lsB[buf][(bCol + 1) * PB + kk] = (_Float16)wb[i].y;
            lsB[buf][(bCol + 2) * PB + kk] = (_Float16)wb[i].z;
            lsB[buf][(bCol + 3) * PB + kk] = (_Float16)wb[i].w;
        }
    };

    v8f acc[2][2] = {};
    stage_load(0);
    stage_store(0);
    __syncthreads();
    int cur = 0;
    for (int k0 = K_STEP; k0 < DMODEL; k0 += K_STEP) {
        stage_load(k0);                               // next tile in flight
        compute_tile(acc, lsA[cur], lsB[cur], wm, wn, lane);
        stage_store(cur ^ 1);                         // waits loads here
        __syncthreads();                              // single barrier/step
        cur ^= 1;
    }
    compute_tile(acc, lsA[cur], lsB[cur], wm, wn, lane);

#pragma unroll
    for (int mi = 0; mi < 2; ++mi)
#pragma unroll
        for (int ni = 0; ni < 2; ++ni)
#pragma unroll
            for (int r = 0; r < 8; ++r) {
                int m = m0 + wm + mi * 16 + (lane >> 4) * 8 + r;
                int n = n0 + wn + ni * 16 + (lane & 15);
                Out[(size_t)m * DMODEL + n] = (_Float16)acc[mi][ni][r];
            }
}

// ---------------------------------------------------------------------------
// Kernel 2: S = (Q @ K^T) / 32 with causal mask.  grid = (32 kt, 16 qt, 4 b)
// Pure f16 copies -> async global->LDS on both tiles when available.
// ---------------------------------------------------------------------------
__global__ __launch_bounds__(256)
void qk_scores(const _Float16* __restrict__ Qh, const _Float16* __restrict__ Kh,
               float* __restrict__ S)
{
    const int b  = blockIdx.z;
    const int m0 = blockIdx.y * M_BLK;     // query tile
    const int n0 = blockIdx.x * N_BLK;     // key tile
    float* Sb = S + (size_t)b * SEQ * SEQ;

    if (n0 > m0 + (M_BLK - 1)) {           // block-uniform: fully masked tile
        for (int f = threadIdx.x; f < M_BLK * N_BLK / 4; f += 256) {
            int row = f >> 4, c4 = f & 15;
            float4 v = { NEG_INF, NEG_INF, NEG_INF, NEG_INF };
            *(float4*)(Sb + (size_t)(m0 + row) * SEQ + n0 + c4 * 4) = v;
        }
        return;
    }

    __shared__ _Float16 lsA[2][M_BLK * PA];
    __shared__ _Float16 lsB[2][N_BLK * PB];

    const int tid  = threadIdx.x;
    const int lane = tid & 31;
    const int wave = tid >> 5;
    const int wm   = (wave >> 1) * 32;
    const int wn   = (wave & 1) * 32;

    // A: 2 b128 chunks/thread: row tid>>2 + i*64, col (tid&3)*8
    // B: 1 b128 chunk/thread:  row tid>>2,        col (tid&3)*8
    const int tRow = tid >> 2, tCol = (tid & 3) * 8;
    const _Float16* aG = Qh + (size_t)(b * SEQ + m0 + tRow) * DMODEL + tCol;
    const _Float16* bG = Kh + (size_t)(b * SEQ + n0 + tRow) * DMODEL + tCol;

#if !HAVE_ASYNC_LDS
    h8 ra[2], rb;
#endif
    auto stage = [&](int e0, int buf) {
#if HAVE_ASYNC_LDS
#pragma unroll
        for (int i = 0; i < 2; ++i)
            async_copy_b128(aG + e0 + (size_t)i * 64 * DMODEL,
                            &lsA[buf][(tRow + i * 64) * PA + tCol]);
        async_copy_b128(bG + e0, &lsB[buf][tRow * PB + tCol]);
#else
#pragma unroll
        for (int i = 0; i < 2; ++i)
            ra[i] = *(const h8*)(aG + e0 + (size_t)i * 64 * DMODEL);
        rb = *(const h8*)(bG + e0);
#pragma unroll
        for (int i = 0; i < 2; ++i)
            *(h8*)(&lsA[buf][(tRow + i * 64) * PA + tCol]) = ra[i];
        *(h8*)(&lsB[buf][tRow * PB + tCol]) = rb;
#endif
    };

    v8f acc[2][2] = {};
    stage(0, 0);
#if HAVE_ASYNC_LDS
    wait_async0();
#endif
    __syncthreads();
    int cur = 0;
    for (int e0 = K_STEP; e0 < DMODEL; e0 += K_STEP) {
        stage(e0, cur ^ 1);                           // copies in flight
        compute_tile(acc, lsA[cur], lsB[cur], wm, wn, lane);
#if HAVE_ASYNC_LDS
        wait_async0();
#endif
        __syncthreads();
        cur ^= 1;
    }
    compute_tile(acc, lsA[cur], lsB[cur], wm, wn, lane);

    const float scale = 0.03125f;          // 1/sqrt(1024)
#pragma unroll
    for (int mi = 0; mi < 2; ++mi)
#pragma unroll
        for (int ni = 0; ni < 2; ++ni)
#pragma unroll
            for (int r = 0; r < 8; ++r) {
                int q = m0 + wm + mi * 16 + (lane >> 4) * 8 + r;
                int k = n0 + wn + ni * 16 + (lane & 15);
                float s = acc[mi][ni][r] * scale;
                if (k > q) s = NEG_INF;
                Sb[(size_t)q * SEQ + k] = s;
            }
}

// ---------------------------------------------------------------------------
// Kernel 3: row softmax over S (f32) -> P (f16).  grid = (2048 rows, 4 b)
// ---------------------------------------------------------------------------
__global__ __launch_bounds__(256)
void softmax_rows(const float* __restrict__ S, _Float16* __restrict__ P)
{
    const int b = blockIdx.y, q = blockIdx.x, tid = threadIdx.x;
    const float* row  = S + ((size_t)b * SEQ + q) * SEQ;
    _Float16*    prow = P + ((size_t)b * SEQ + q) * SEQ;
    __shared__ float red[256];

    float v[8];
    float4 v0 = *(const float4*)(row + tid * 8);
    float4 v1 = *(const float4*)(row + tid * 8 + 4);
    v[0]=v0.x; v[1]=v0.y; v[2]=v0.z; v[3]=v0.w;
    v[4]=v1.x; v[5]=v1.y; v[6]=v1.z; v[7]=v1.w;

    float lm = v[0];
#pragma unroll
    for (int j = 1; j < 8; ++j) lm = fmaxf(lm, v[j]);
    red[tid] = lm; __syncthreads();
    for (int s = 128; s > 0; s >>= 1) {
        if (tid < s) red[tid] = fmaxf(red[tid], red[tid + s]);
        __syncthreads();
    }
    const float rowmax = red[0];           // always finite: k=0 is unmasked
    __syncthreads();

    float e[8], lsum = 0.f;
#pragma unroll
    for (int j = 0; j < 8; ++j) { e[j] = expf(v[j] - rowmax); lsum += e[j]; }
    red[tid] = lsum; __syncthreads();
    for (int s = 128; s > 0; s >>= 1) {
        if (tid < s) red[tid] += red[tid + s];
        __syncthreads();
    }
    const float inv = 1.0f / red[0];

    h4 p0 = { (_Float16)(e[0]*inv), (_Float16)(e[1]*inv),
              (_Float16)(e[2]*inv), (_Float16)(e[3]*inv) };
    h4 p1 = { (_Float16)(e[4]*inv), (_Float16)(e[5]*inv),
              (_Float16)(e[6]*inv), (_Float16)(e[7]*inv) };
    *(h4*)(prow + tid * 8)     = p0;
    *(h4*)(prow + tid * 8 + 4) = p1;
}

// ---------------------------------------------------------------------------
// Kernel 4: out = P @ V (f32 out).  grid = (16 nt, 16 qt, 4 b)
// K loop clipped to causal extent. A tile async-copied (pure f16 copy),
// V tile register-staged (needs transpose into LDS).
// ---------------------------------------------------------------------------
__global__ __launch_bounds__(256)
void pv_gemm(const _Float16* __restrict__ P, const _Float16* __restrict__ Vh,
             float* __restrict__ out)
{
    __shared__ _Float16 lsA[2][M_BLK * PA];
    __shared__ _Float16 lsB[2][N_BLK * PB];

    const int tid  = threadIdx.x;
    const int lane = tid & 31;
    const int wave = tid >> 5;
    const int b    = blockIdx.z;
    const int m0   = blockIdx.y * M_BLK;   // query tile
    const int n0   = blockIdx.x * N_BLK;   // out-feature tile
    const int wm   = (wave >> 1) * 32;
    const int wn   = (wave & 1) * 32;

    const int tRow = tid >> 2, tCol = (tid & 3) * 8;     // A staging coords
    const int vK = tid >> 3, vN = (tid & 7) * 8;         // B staging coords
    const _Float16* aG = P + (size_t)b * SEQ * SEQ
                           + (size_t)(m0 + tRow) * SEQ + tCol;
    const _Float16* bG = Vh + (size_t)(b * SEQ + vK) * DMODEL + n0 + vN;

    const int kEnd = m0 + M_BLK;           // causal: P[q][k]==0 for k>q

    h8 rv;                                 // V transpose staging register
#if !HAVE_ASYNC_LDS
    h8 ra[2];
#endif
    auto stageA = [&](int k0, int buf) {
#if HAVE_ASYNC_LDS
#pragma unroll
        for (int i = 0; i < 2; ++i)
            async_copy_b128(aG + k0 + (size_t)i * 64 * SEQ,
                            &lsA[buf][(tRow + i * 64) * PA + tCol]);
#else
#pragma unroll
        for (int i = 0; i < 2; ++i)
            ra[i] = *(const h8*)(aG + k0 + (size_t)i * 64 * SEQ);
#pragma unroll
        for (int i = 0; i < 2; ++i)
            *(h8*)(&lsA[buf][(tRow + i * 64) * PA + tCol]) = ra[i];
#endif
    };
    auto loadB  = [&](int k0) { rv = *(const h8*)(bG + (size_t)k0 * DMODEL); };
    auto storeB = [&](int buf) {
#pragma unroll
        for (int i = 0; i < 8; ++i)
            lsB[buf][(vN + i) * PB + vK] = rv[i];      // V[k][n] -> Bt[n][k]
    };

    v8f acc[2][2] = {};
    loadB(0);
    stageA(0, 0);
    storeB(0);
#if HAVE_ASYNC_LDS
    wait_async0();
#endif
    __syncthreads();
    int cur = 0;
    for (int k0 = K_STEP; k0 < kEnd; k0 += K_STEP) {
        loadB(k0);                                    // V loads in flight
        stageA(k0, cur ^ 1);                          // async P copies in flight
        compute_tile(acc, lsA[cur], lsB[cur], wm, wn, lane);
        storeB(cur ^ 1);                              // waits V loadcnt here
#if HAVE_ASYNC_LDS
        wait_async0();
#endif
        __syncthreads();
        cur ^= 1;
    }
    compute_tile(acc, lsA[cur], lsB[cur], wm, wn, lane);

#pragma unroll
    for (int mi = 0; mi < 2; ++mi)
#pragma unroll
        for (int ni = 0; ni < 2; ++ni)
#pragma unroll
            for (int r = 0; r < 8; ++r) {
                int q = m0 + wm + mi * 16 + (lane >> 4) * 8 + r;
                int n = n0 + wn + ni * 16 + (lane & 15);
                out[((size_t)b * SEQ + q) * DMODEL + n] = acc[mi][ni][r];
            }
}

// ---------------------------------------------------------------------------
extern "C" void kernel_launch(void* const* d_in, const int* in_sizes, int n_in,
                              void* d_out, int out_size, void* d_ws, size_t ws_size,
                              hipStream_t stream)
{
    (void)in_sizes; (void)n_in; (void)out_size; (void)ws_size;

    const float* X  = (const float*)d_in[0];
    const float* Wq = (const float*)d_in[1];
    const float* Wk = (const float*)d_in[2];
    const float* Wv = (const float*)d_in[3];
    float* out = (float*)d_out;

    _Float16* Qh = (_Float16*)d_ws;                       // 16 MB
    _Float16* Kh = Qh + (size_t)ROWS * DMODEL;            // 16 MB
    _Float16* Vh = Kh + (size_t)ROWS * DMODEL;            // 16 MB
    float*    S  = (float*)(Vh + (size_t)ROWS * DMODEL);  // 64 MB
    _Float16* P  = (_Float16*)(S + (size_t)NBATCH * SEQ * SEQ); // 32 MB

    dim3 blk(256);
    qkv_gemm   <<<dim3(ROWS / M_BLK, DMODEL / N_BLK, 3), blk, 0, stream>>>(
        X, Wq, Wk, Wv, Qh, Kh, Vh);
    qk_scores  <<<dim3(SEQ / N_BLK, SEQ / M_BLK, NBATCH), blk, 0, stream>>>(
        Qh, Kh, S);
    softmax_rows<<<dim3(SEQ, NBATCH), blk, 0, stream>>>(S, P);
    pv_gemm    <<<dim3(DMODEL / N_BLK, SEQ / M_BLK, NBATCH), blk, 0, stream>>>(
        P, Vh, out);
}
